// RhnLinTCTied_53231824667306
// MI455X (gfx1250) — matrix operand order repaired and checked by reading, not compiled, not run on hardware
//
#include <hip/hip_runtime.h>
#include <hip/hip_bf16.h>
#include <cstdint>

#define BB  64
#define TT  512
#define INN 1024
#define HH  1024
#define LL  5

typedef __bf16 bf16_t;
typedef __attribute__((ext_vector_type(16))) __bf16 v16bf;
typedef __attribute__((ext_vector_type(8)))  __bf16 v8bf;
typedef __attribute__((ext_vector_type(8)))  float  v8f;

// ---------------------------------------------------------------------------
// WMMA helpers (bf16 A/B, f32 accumulate) — native CDNA5 matrix path.
// ---------------------------------------------------------------------------
__device__ __forceinline__ v8f wmma_bf16(v16bf a, v16bf b, v8f c) {
  // (neg_a, A, neg_b, B, c_mod, C, reuse_a, reuse_b)
  return __builtin_amdgcn_wmma_f32_16x16x32_bf16(false, a, false, b, (short)0, c,
                                                 false, false);
}

// A-matrix 16x32 bf16 tile. ISA layout: lane (m = lane%16) holds K=off..off+7
// in halves 0..7 and K=16+off..23+off in halves 8..15, off = 8*(lane>=16).
__device__ __forceinline__ v16bf load_a_tile(const bf16_t* __restrict__ A, int lda,
                                             int m0, int k0, int lane) {
  const int m   = m0 + (lane & 15);
  const int off = (lane >> 4) << 3;
  const bf16_t* p = A + (size_t)m * lda + (k0 + off);
  v8bf lo = *(const v8bf*)(p);
  v8bf hi = *(const v8bf*)(p + 16);
  return __builtin_shufflevector(lo, hi, 0, 1, 2, 3, 4, 5, 6, 7,
                                         8, 9, 10, 11, 12, 13, 14, 15);
}

// B-matrix 32x16 bf16 tile: lane n%16 holds column n, K=off..off+7 in halves
// 0..7 and K=16+off..23+off in halves 8..15, off = 8*(lane>=16). Column n of B
// == row n of the K-major weight array (NT layout — no transpose needed).
__device__ __forceinline__ v16bf load_b_tile(const bf16_t* __restrict__ W, int ldb,
                                             int n0, int k0, int lane) {
  const int n    = n0 + (lane & 15);
  const int koff = (lane >> 4) << 3;
  const bf16_t* p = W + (size_t)n * ldb + (k0 + koff);
  v8bf lo = *(const v8bf*)(p);
  v8bf hi = *(const v8bf*)(p + 16);
  return __builtin_shufflevector(lo, hi, 0, 1, 2, 3, 4, 5, 6, 7,
                                         8, 9, 10, 11, 12, 13, 14, 15);
}

// ---------------------------------------------------------------------------
// Phase 1: conversions
// ---------------------------------------------------------------------------
__global__ void f32_to_bf16_kernel(const float* __restrict__ in,
                                   bf16_t* __restrict__ out, size_t n) {
  size_t i = (size_t)blockIdx.x * blockDim.x + threadIdx.x;
  if (i < n) out[i] = (bf16_t)in[i];
}

// out[c][r] = in[r][c]  (out shape [cols][rows], K-major for WMMA B)
__global__ void transpose_f32_to_bf16_kernel(const float* __restrict__ in,
                                             bf16_t* __restrict__ out,
                                             int rows, int cols) {
  size_t i = (size_t)blockIdx.x * blockDim.x + threadIdx.x;
  size_t n = (size_t)rows * cols;
  if (i < n) {
    int cc = (int)(i / rows);
    int rr = (int)(i % rows);
    out[i] = (bf16_t)in[(size_t)rr * cols + cc];
  }
}

__global__ void init_state_kernel(const float* __restrict__ s_in,
                                  float* __restrict__ s0,
                                  bf16_t* __restrict__ sb0,
                                  unsigned* __restrict__ bar) {
  size_t i = (size_t)blockIdx.x * blockDim.x + threadIdx.x;
  if (i == 0) *bar = 0u;  // re-arm grid barrier every launch (graph-replay safe)
  if (i < (size_t)BB * HH) {
    float v = s_in[i];
    s0[i]  = v;
    sb0[i] = (bf16_t)v;
  }
}

// ---------------------------------------------------------------------------
// Phase 2: input projections whx/wtx for all B*T rows (parallel GEMM).
// 2048 row-tiles x 32 column groups (4 tiles each, wh|wt). Each wave owns a
// 16x64 output strip: one A fragment feeds 4 WMMAs per k-step (4x A reuse).
// ---------------------------------------------------------------------------
__global__ void __launch_bounds__(256)
proj_gemm_kernel(const bf16_t* __restrict__ Xbf, const bf16_t* __restrict__ whT,
                 const bf16_t* __restrict__ wtT, float* __restrict__ whx,
                 float* __restrict__ wtx) {
  const int lane = threadIdx.x & 31;
  const int wave = threadIdx.x >> 5;
  const int g    = blockIdx.x * 8 + wave;
  const int mt   = g & ((BB * TT / 16) - 1);  // 0..2047
  const int ng   = g >> 11;                   // 0..31 column groups of 64
  const int m0   = mt << 4;

  const bf16_t* W;
  float* D;
  int n0;
  if (ng < (HH / 64)) { W = whT; D = whx; n0 = ng << 6; }
  else                { W = wtT; D = wtx; n0 = (ng - HH / 64) << 6; }

  v8f c0 = {}, c1 = {}, c2 = {}, c3 = {};
  for (int k0 = 0; k0 < INN; k0 += 32) {
    v16bf a  = load_a_tile(Xbf, INN, m0, k0, lane);
    v16bf b0 = load_b_tile(W, INN, n0,      k0, lane);
    v16bf b1 = load_b_tile(W, INN, n0 + 16, k0, lane);
    v16bf b2 = load_b_tile(W, INN, n0 + 32, k0, lane);
    v16bf b3 = load_b_tile(W, INN, n0 + 48, k0, lane);
    c0 = wmma_bf16(a, b0, c0);
    c1 = wmma_bf16(a, b1, c1);
    c2 = wmma_bf16(a, b2, c2);
    c3 = wmma_bf16(a, b3, c3);
  }
  const int rowoff = (lane >> 4) << 3;
  const int coll   = lane & 15;
#pragma unroll
  for (int r = 0; r < 8; ++r) {
    const size_t row = (size_t)(m0 + r + rowoff) * HH;  // row = b*TT + t
    D[row + n0      + coll] = c0[r];
    D[row + n0 + 16 + coll] = c1[r];
    D[row + n0 + 32 + coll] = c2[r];
    D[row + n0 + 48 + coll] = c3[r];
  }
}

// ---------------------------------------------------------------------------
// Phase 3: persistent recurrent scan. 64 blocks x 8 waves (spread across 64
// WGPs for aggregate L2/LDS bandwidth); waves paired per 16x16 state tile
// (even wave: pre_h GEMM, odd wave: pre_t GEMM -> LDS handoff).
// Weights (20MB bf16) stay L2-resident across all 2560 steps.
// ---------------------------------------------------------------------------
#define SCAN_NWG 64

__global__ void __launch_bounds__(256)
rhn_scan_kernel(const bf16_t* __restrict__ rhw_bf, const bf16_t* __restrict__ rtw_bf,
                const float* __restrict__ rh_b, const float* __restrict__ rt_b,
                const float* __restrict__ whx, const float* __restrict__ wtx,
                float* __restrict__ s_a, float* __restrict__ s_b,
                bf16_t* __restrict__ sb_a, bf16_t* __restrict__ sb_b,
                float* __restrict__ out, unsigned* __restrict__ bar) {
  __shared__ float lds_t[4][256];

  const int lane   = threadIdx.x & 31;
  const int wave   = threadIdx.x >> 5;
  const int pair   = wave >> 1;              // 0..3
  const int is_t   = wave & 1;               // wave-uniform: EXEC stays full
  const int P      = blockIdx.x * 4 + pair;  // 0..255 tile pairs
  const int m0     = (P & 3) << 4;           // 4 row tiles over B=64
  const int n0     = (P >> 2) << 4;          // 64 col tiles over H=1024
  const int colg   = n0 + (lane & 15);
  const int rowoff = (lane >> 4) << 3;

  int t = 0, l = 0;
  for (int it = 0; it < TT * LL; ++it) {
    const float*  scur  = (it & 1) ? s_b : s_a;
    float*        snext = (it & 1) ? s_a : s_b;
    const bf16_t* sbc   = (it & 1) ? sb_b : sb_a;
    bf16_t*       sbn   = (it & 1) ? sb_a : sb_b;

    const bf16_t* W  = (is_t ? rtw_bf : rhw_bf) + (size_t)l * HH * HH;
    const float*  bi = (is_t ? rt_b : rh_b) + l * HH;
    const float*  pj = is_t ? wtx : whx;

    v8f c = {};
    for (int k0 = 0; k0 < HH; k0 += 32) {
      v16bf a = load_a_tile(sbc, HH, m0, k0, lane);
      v16bf b = load_b_tile(W, HH, n0, k0, lane);
      __builtin_prefetch(W + (size_t)colg * HH + k0 + 32, 0, 1);
      c = wmma_bf16(a, b, c);
    }

    const float bg = bi[colg];
    if (is_t) {
#pragma unroll
      for (int r = 0; r < 8; ++r) {
        const int b_ = m0 + r + rowoff;
        float pre = c[r] + bg;
        if (l == 0) pre += pj[((size_t)b_ * TT + t) * HH + colg];
        lds_t[pair][r * 32 + lane] = pre;
      }
    }
    __syncthreads();
    if (!is_t) {
#pragma unroll
      for (int r = 0; r < 8; ++r) {
        const int b_ = m0 + r + rowoff;
        float pre_h = c[r] + bg;
        if (l == 0) pre_h += pj[((size_t)b_ * TT + t) * HH + colg];
        const float pre_t = lds_t[pair][r * 32 + lane];
        const size_t sidx = (size_t)b_ * HH + colg;
        const float so = scur[sidx];
        const float hv = tanhf(pre_h);
        const float tg = 1.0f / (1.0f + __expf(-pre_t));
        const float sn = (hv - so) * tg + so;
        snext[sidx] = sn;
        sbn[sidx]   = (bf16_t)sn;
        if (l == LL - 1) {
          out[((size_t)b_ * TT + t) * HH + colg] = sn;             // outputs[b][t][g]
          if (t == TT - 1) out[(size_t)BB * TT * HH + sidx] = sn;  // s_final
        }
      }
    }

    // ---- grid-wide barrier (monotonic counter, release/acquire) ----
    __threadfence();
    __syncthreads();
    if (threadIdx.x == 0) {
      __hip_atomic_fetch_add(bar, 1u, __ATOMIC_RELEASE, __HIP_MEMORY_SCOPE_AGENT);
      const unsigned target = (unsigned)(it + 1) * gridDim.x;
      while (__hip_atomic_load(bar, __ATOMIC_ACQUIRE, __HIP_MEMORY_SCOPE_AGENT) < target)
        __builtin_amdgcn_s_sleep(1);
    }
    __syncthreads();
    __threadfence();

    if (++l == LL) { l = 0; ++t; }
  }
}

// ---------------------------------------------------------------------------
extern "C" void kernel_launch(void* const* d_in, const int* in_sizes, int n_in,
                              void* d_out, int out_size, void* d_ws, size_t ws_size,
                              hipStream_t stream) {
  (void)in_sizes; (void)n_in; (void)out_size; (void)ws_size;

  const float* x    = (const float*)d_in[0];   // [B,T,IN]
  const float* s_in = (const float*)d_in[1];   // [B,H]
  const float* w_h  = (const float*)d_in[2];   // [IN,H]
  const float* w_t  = (const float*)d_in[3];   // [IN,H]
  const float* rh_w = (const float*)d_in[4];   // [L,H,H]
  const float* rh_b = (const float*)d_in[5];   // [L,H]
  const float* rt_w = (const float*)d_in[6];   // [L,H,H]
  const float* rt_b = (const float*)d_in[7];   // [L,H]
  float* out = (float*)d_out;                  // [B,T,H] ++ [B,H]

  char* ws = (char*)d_ws;
  size_t off = 0;
  auto carve = [&](size_t bytes) -> char* {
    char* p = ws + off;
    off += (bytes + 255) & ~(size_t)255;
    return p;
  };
  unsigned* bar   = (unsigned*)carve(256);
  bf16_t*   x_bf  = (bf16_t*)carve((size_t)BB * TT * INN * 2);
  bf16_t*   whT   = (bf16_t*)carve((size_t)INN * HH * 2);
  bf16_t*   wtT   = (bf16_t*)carve((size_t)INN * HH * 2);
  bf16_t*   rh_bf = (bf16_t*)carve((size_t)LL * HH * HH * 2);
  bf16_t*   rt_bf = (bf16_t*)carve((size_t)LL * HH * HH * 2);
  float*    whx   = (float*)carve((size_t)BB * TT * HH * 4);
  float*    wtx   = (float*)carve((size_t)BB * TT * HH * 4);
  float*    s0    = (float*)carve((size_t)BB * HH * 4);
  float*    s1    = (float*)carve((size_t)BB * HH * 4);
  bf16_t*   sb0   = (bf16_t*)carve((size_t)BB * HH * 2);
  bf16_t*   sb1   = (bf16_t*)carve((size_t)BB * HH * 2);

  // Phase 1: conversions (stream-ordered).
  {
    size_t n = (size_t)BB * TT * INN;
    f32_to_bf16_kernel<<<(unsigned)((n + 255) / 256), 256, 0, stream>>>(x, x_bf, n);
  }
  {
    size_t n = (size_t)INN * HH;
    transpose_f32_to_bf16_kernel<<<(unsigned)((n + 255) / 256), 256, 0, stream>>>(
        w_h, whT, INN, HH);
    transpose_f32_to_bf16_kernel<<<(unsigned)((n + 255) / 256), 256, 0, stream>>>(
        w_t, wtT, INN, HH);
  }
  {
    size_t n = (size_t)LL * HH * HH;
    f32_to_bf16_kernel<<<(unsigned)((n + 255) / 256), 256, 0, stream>>>(rh_w, rh_bf, n);
    f32_to_bf16_kernel<<<(unsigned)((n + 255) / 256), 256, 0, stream>>>(rt_w, rt_bf, n);
  }
  init_state_kernel<<<(BB * HH + 255) / 256, 256, 0, stream>>>(s_in, s0, sb0, bar);

  // Phase 2: input projections (2048 row-tiles x 32 col-groups / 8 waves).
  proj_gemm_kernel<<<(BB * TT / 16) * (2 * HH / 64) / 8, 256, 0, stream>>>(
      x_bf, whT, wtT, whx, wtx);

  // Phase 3: persistent recurrent scan.
  rhn_scan_kernel<<<SCAN_NWG, 256, 0, stream>>>(rh_bf, rt_bf, rh_b, rt_b, whx, wtx,
                                                s0, s1, sb0, sb1, out, bar);
}
